// RowColumnAttention_69939247448366
// MI455X (gfx1250) — matrix operand order repaired and checked by reading, not compiled, not run on hardware
//
#include <hip/hip_runtime.h>

// ---------------- problem constants ----------------
#define BSZ   2
#define SLEN  2048
#define HDIM  1024
#define NH    16
#define KVH   4
#define HD    64
#define MROWS (BSZ * SLEN)       // 4096
#define QKVN  (NH*HD + 2*KVH*HD) // 1536

typedef __attribute__((ext_vector_type(16))) __bf16 v16bf;
typedef __attribute__((ext_vector_type(8)))  __bf16 v8bf;
typedef __attribute__((ext_vector_type(8)))  float  v8f;

// ---- CDNA5 async copy: global -> LDS, 16 bytes per lane (ASYNCcnt-tracked) ----
__device__ __forceinline__ void async_b128(unsigned lds_off, const void* gaddr) {
    asm volatile("global_load_async_to_lds_b128 %0, %1, off"
                 :: "v"(lds_off), "v"(gaddr) : "memory");
}
__device__ __forceinline__ void wait_async0() {
    asm volatile("s_wait_asynccnt 0x0" ::: "memory");
}

// ---------------- f32 -> bf16 convert ----------------
__global__ void f32_to_bf16_k(const float* __restrict__ in, __bf16* __restrict__ out, int n) {
    int i = blockIdx.x * blockDim.x + threadIdx.x;
    if (i < n) out[i] = (__bf16)in[i];
}

// ---------------- GEMM: C[M,N] = A[M,K] * B[N,K]^T (bf16 in, f32 out) ----------------
// Register-blocked: each wave computes a 32x64 tile (2 M-subtiles x 4 N-subtiles,
// 8 WMMAs per 32-wide K step). 4 waves/block -> 32x256 block tile.
__global__ __launch_bounds__(128) void gemm_bf16_nt(
    const __bf16* __restrict__ A, const __bf16* __restrict__ Bm,
    float* __restrict__ C, int M, int N, int K)
{
    const int wave = threadIdx.x >> 5;
    const int lane = threadIdx.x & 31;
    const int g    = lane >> 4;     // half-wave id
    const int hn   = lane & 15;     // m / n within a 16-subtile
    const int n0   = (blockIdx.x * 4 + wave) * 64;  // wave's first column
    const int m0   = blockIdx.y * 32;               // wave's first row

    const __bf16* arow[2];
    arow[0] = A + (size_t)(m0 + hn) * K;
    arow[1] = A + (size_t)(m0 + 16 + hn) * K;
    const __bf16* brow[4];
#pragma unroll
    for (int ns = 0; ns < 4; ++ns)
        brow[ns] = Bm + (size_t)(n0 + ns * 16 + hn) * K;

    v8f acc[2][4];
#pragma unroll
    for (int ms = 0; ms < 2; ++ms)
#pragma unroll
        for (int ns = 0; ns < 4; ++ns)
            acc[ms][ns] = (v8f){0.f,0.f,0.f,0.f,0.f,0.f,0.f,0.f};

    for (int k0 = 0; k0 < K; k0 += 32) {
        v16bf af[2];
#pragma unroll
        for (int ms = 0; ms < 2; ++ms) {
            // A fragment: elem e -> K = k0 + g*16 + e (contiguous)
            union { v16bf v; v8bf h8[2]; } ua;
            ua.h8[0] = *(const v8bf*)(arow[ms] + k0 + g * 16);
            ua.h8[1] = *(const v8bf*)(arow[ms] + k0 + g * 16 + 8);
            af[ms] = ua.v;
        }
        v16bf bfr[4];
#pragma unroll
        for (int ns = 0; ns < 4; ++ns)  // B fragment: K = k0 + g*16 + e (contiguous)
            bfr[ns] = *(const v16bf*)(brow[ns] + k0 + g * 16);

        __builtin_prefetch(arow[0] + k0 + 64, 0, 0);
        __builtin_prefetch(brow[0] + k0 + 64, 0, 0);

#pragma unroll
        for (int ms = 0; ms < 2; ++ms)
#pragma unroll
            for (int ns = 0; ns < 4; ++ns)
                acc[ms][ns] = __builtin_amdgcn_wmma_f32_16x16x32_bf16(
                    false, af[ms], false, bfr[ns], (short)0, acc[ms][ns], false, false);
    }

    // C layout: vgpr r -> row m0 + ms*16 + r + 8*g, col n0 + ns*16 + hn
#pragma unroll
    for (int ms = 0; ms < 2; ++ms) {
        float* cp = C + (size_t)(m0 + ms * 16 + 8 * g) * N + n0 + hn;
#pragma unroll
        for (int ns = 0; ns < 4; ++ns)
#pragma unroll
            for (int r = 0; r < 8; ++r)
                cp[(size_t)r * N + ns * 16] = acc[ms][ns][r];
    }
}

// ---------------- QKV post-process: RMS-norm + RoPE + relayout ----------------
__global__ __launch_bounds__(128) void qkv_post_k(
    const float* __restrict__ QKV,   // [B*S][1536]
    const float* __restrict__ cosb,  // [B][S][64]
    const float* __restrict__ sinb,  // [B][S][64]
    const float* __restrict__ qnw, const float* __restrict__ knw,
    __bf16* __restrict__ Qo,   // [B][NH][S][64]
    __bf16* __restrict__ Ko,   // [B][KVH][S][64]
    __bf16* __restrict__ Vt)   // [B][KVH][64][S]  (transposed)
{
    const int wave = threadIdx.x >> 5;
    const int lane = threadIdx.x & 31;
    const int row  = blockIdx.x * 4 + wave;      // 0 .. B*S*24-1
    const int hk   = row % 24;
    const int bs   = row / 24;
    const int s    = bs % SLEN;
    const int b    = bs / SLEN;

    int off;
    if (hk < 16)      off = hk * 64;
    else if (hk < 20) off = 1024 + (hk - 16) * 64;
    else              off = 1280 + (hk - 20) * 64;

    const float* x = QKV + (size_t)bs * QKVN + off;
    float x0 = x[lane];
    float x1 = x[lane + 32];

    if (hk < 20) {
        float ss = x0 * x0 + x1 * x1;
        ss += __shfl_xor(ss, 1, 32);
        ss += __shfl_xor(ss, 2, 32);
        ss += __shfl_xor(ss, 4, 32);
        ss += __shfl_xor(ss, 8, 32);
        ss += __shfl_xor(ss, 16, 32);
        float inv = rsqrtf(ss * (1.0f / 64.0f) + 1e-6f);
        const float* w = (hk < 16) ? qnw : knw;
        x0 = x0 * inv * w[lane];
        x1 = x1 * inv * w[lane + 32];
        const float* cb = cosb + ((size_t)b * SLEN + s) * 64;
        const float* sb = sinb + ((size_t)b * SLEN + s) * 64;
        float r0 = x0 * cb[lane]      - x1 * sb[lane];
        float r1 = x1 * cb[lane + 32] + x0 * sb[lane + 32];
        x0 = r0; x1 = r1;
    }

    if (hk < 16) {
        __bf16* dst = Qo + (((size_t)b * NH + hk) * SLEN + s) * 64;
        dst[lane] = (__bf16)x0; dst[lane + 32] = (__bf16)x1;
    } else if (hk < 20) {
        __bf16* dst = Ko + (((size_t)b * KVH + (hk - 16)) * SLEN + s) * 64;
        dst[lane] = (__bf16)x0; dst[lane + 32] = (__bf16)x1;
    } else {
        __bf16* dst = Vt + ((size_t)b * KVH + (hk - 20)) * 64 * SLEN;
        dst[(size_t)lane * SLEN + s]        = (__bf16)x0;
        dst[(size_t)(lane + 32) * SLEN + s] = (__bf16)x1;
    }
}

// ---- cooperative async staging of one 32-key K tile + V^T tile into LDS ----
__device__ __forceinline__ void stage_kv(
    __bf16 (*kb)[64], __bf16 (*vb)[32],
    const __bf16* kbase, const __bf16* vbase, int j0, int tid)
{
    // K: 32 rows x 64 cols (4 KB) -> 256 x 16B chunks, 2 per thread
#pragma unroll
    for (int i = 0; i < 2; ++i) {
        int c = tid + i * 128;
        int r = c >> 3, co = (c & 7) * 8;
        async_b128((unsigned)(unsigned long long)&kb[r][co],
                   kbase + (size_t)(j0 + r) * HD + co);
    }
    // V^T: 64 rows x 32 cols (4 KB) -> 256 x 16B chunks, 2 per thread
#pragma unroll
    for (int i = 0; i < 2; ++i) {
        int c = tid + i * 128;
        int d = c >> 2, co = (c & 3) * 8;
        async_b128((unsigned)(unsigned long long)&vb[d][co],
                   vbase + (size_t)d * SLEN + j0 + co);
    }
}

// ---------------- Flash attention (causal, GQA) ----------------
// Block = 4 waves sharing (b,h); wave w owns q-rows [q0,q0+16).
// K/V tiles (32 keys) staged once per block into double-buffered LDS via
// global_load_async_to_lds_b128, overlapped with WMMA compute.
__global__ __launch_bounds__(128) void flash_attn_k(
    const __bf16* __restrict__ Q,   // [B][NH][S][64]
    const __bf16* __restrict__ Kc,  // [B][KVH][S][64]
    const __bf16* __restrict__ Vt,  // [B][KVH][64][S]
    __bf16* __restrict__ O)         // [B][S][NH*64]
{
    __shared__ __bf16 kbuf[2][32][64];   // 8 KB double-buffered K tile
    __shared__ __bf16 vbuf[2][64][32];   // 8 KB double-buffered V^T tile
    __shared__ __bf16 pbuf[4][16][32];   // per-wave P staging (C->A layout)

    const int tid  = threadIdx.x;
    const int wave = tid >> 5;
    const int lane = tid & 31;
    const int g    = lane >> 4;
    const int hn   = lane & 15;
    const int h    = blockIdx.y;
    const int b    = blockIdx.z;
    const int kvh  = h / (NH / KVH);
    const int q0   = blockIdx.x * 64 + wave * 16;

    const __bf16* qbase = Q  + (((size_t)b * NH + h)    * SLEN + q0) * HD;
    const __bf16* kbase = Kc + (((size_t)b * KVH + kvh) * SLEN) * HD;
    const __bf16* vbase = Vt + (((size_t)b * KVH + kvh) * HD) * SLEN;

    // Q fragments (A-layout) for d-chunks 0 and 32, loaded once.
    v16bf qa[2];
    {
        const __bf16* qrow = qbase + (size_t)hn * HD;
#pragma unroll
        for (int dc = 0; dc < 2; ++dc) {
            union { v16bf v; v8bf h8[2]; } u;
            u.h8[0] = *(const v8bf*)(qrow + dc * 32 + g * 16);
            u.h8[1] = *(const v8bf*)(qrow + dc * 32 + g * 16 + 8);
            qa[dc] = u.v;
        }
    }

    v8f o_acc[4];
#pragma unroll
    for (int i = 0; i < 4; ++i) o_acc[i] = (v8f){0.f,0.f,0.f,0.f,0.f,0.f,0.f,0.f};
    float m_run[8], l_run[8];
#pragma unroll
    for (int r = 0; r < 8; ++r) { m_run[r] = -1e30f; l_run[r] = 0.f; }

    const int jend  = q0 + 16;                 // this wave's causal bound (exclusive)
    const int nIter = (blockIdx.x + 1) * 2;    // block-uniform trip count (32 keys/iter)

    stage_kv(kbuf[0], vbuf[0], kbase, vbase, 0, tid);   // prologue

    for (int it = 0; it < nIter; ++it) {
        const int j0  = it * 32;
        const int buf = it & 1;

        wait_async0();        // own stage(it) loads complete
        __syncthreads();      // all waves' stage(it) complete; compute(it-1) done

        if (it + 1 < nIter)   // overlap next tile's DMA with this tile's WMMAs
            stage_kv(kbuf[buf ^ 1], vbuf[buf ^ 1], kbase, vbase, j0 + 32, tid);

        if (j0 < jend) {      // wave-uniform guard -> EXEC stays all-ones
            // ---- scores: two 16-key tiles, each 2 WMMAs over HD=64 ----
            v8f sc[2];
#pragma unroll
            for (int t = 0; t < 2; ++t) {
                v8f c = (v8f){0.f,0.f,0.f,0.f,0.f,0.f,0.f,0.f};
#pragma unroll
                for (int dc = 0; dc < 2; ++dc) {
                    v16bf kb = *(const v16bf*)&kbuf[buf][t * 16 + hn][dc * 32 + g * 16];
                    c = __builtin_amdgcn_wmma_f32_16x16x32_bf16(false, qa[dc], false, kb,
                                                                (short)0, c, false, false);
                }
                sc[t] = c;
            }

            // ---- mask + online softmax ----
            float p[2][8], chmax[8];
#pragma unroll
            for (int r = 0; r < 8; ++r) chmax[r] = -1e30f;
#pragma unroll
            for (int t = 0; t < 2; ++t) {
                const int j = j0 + t * 16 + hn;
#pragma unroll
                for (int r = 0; r < 8; ++r) {
                    const int qrow = q0 + r + 8 * g;
                    float s = sc[t][r] * 0.125f;
                    s = (j > qrow) ? -1e30f : s;
                    p[t][r] = s;
                    chmax[r] = fmaxf(chmax[r], s);
                }
            }
#pragma unroll
            for (int r = 0; r < 8; ++r) {
                float x = chmax[r];
                x = fmaxf(x, __shfl_xor(x, 1, 32));
                x = fmaxf(x, __shfl_xor(x, 2, 32));
                x = fmaxf(x, __shfl_xor(x, 4, 32));
                x = fmaxf(x, __shfl_xor(x, 8, 32));
                chmax[r] = x;
            }
            float rescale[8];
#pragma unroll
            for (int r = 0; r < 8; ++r) {
                float mnew = fmaxf(m_run[r], chmax[r]);
                rescale[r] = __expf(m_run[r] - mnew);
                m_run[r] = mnew;
            }
#pragma unroll
            for (int t = 0; t < 2; ++t)
#pragma unroll
                for (int r = 0; r < 8; ++r) {
                    float e = __expf(p[t][r] - m_run[r]);
                    p[t][r] = e;
                    pbuf[wave][r + 8 * g][t * 16 + hn] = (__bf16)e;
                }
#pragma unroll
            for (int r = 0; r < 8; ++r) {
                float rs = p[0][r] + p[1][r];
                rs += __shfl_xor(rs, 1, 32);
                rs += __shfl_xor(rs, 2, 32);
                rs += __shfl_xor(rs, 4, 32);
                rs += __shfl_xor(rs, 8, 32);
                l_run[r] = l_run[r] * rescale[r] + rs;
            }

            // ---- reload P in A-layout (same-wave LDS ordering) ----
            union { v16bf v; v8bf h8[2]; } up;
            up.h8[0] = *(const v8bf*)&pbuf[wave][hn][g * 16];
            up.h8[1] = *(const v8bf*)&pbuf[wave][hn][g * 16 + 8];
            const v16bf pa = up.v;

            // ---- O = O*rescale + P @ V ----
#pragma unroll
            for (int dt = 0; dt < 4; ++dt) {
                v8f oo = o_acc[dt];
#pragma unroll
                for (int r = 0; r < 8; ++r) oo[r] *= rescale[r];
                v16bf vb = *(const v16bf*)&vbuf[buf][dt * 16 + hn][g * 16];
                o_acc[dt] = __builtin_amdgcn_wmma_f32_16x16x32_bf16(false, pa, false, vb,
                                                                    (short)0, oo, false, false);
            }
        }
        __syncthreads();  // compute(it) reads done before stage(it+2) overwrites
    }

    // ---- finalize + store bf16 [b][s][h*64+d] ----
    __bf16* obase = O + ((size_t)b * SLEN + q0) * (NH * HD) + h * HD;
#pragma unroll
    for (int dt = 0; dt < 4; ++dt)
#pragma unroll
        for (int r = 0; r < 8; ++r) {
            float val = o_acc[dt][r] / l_run[r];
            obase[(size_t)(r + 8 * g) * (NH * HD) + dt * 16 + hn] = (__bf16)val;
        }
}

// ---------------- launch ----------------
extern "C" void kernel_launch(void* const* d_in, const int* in_sizes, int n_in,
                              void* d_out, int out_size, void* d_ws, size_t ws_size,
                              hipStream_t stream) {
    const float* hidden = (const float*)d_in[0];
    /* d_in[1] attention_mask: pure causal -> computed analytically */
    const float* cosb = (const float*)d_in[2];
    const float* sinb = (const float*)d_in[3];
    const float* qw   = (const float*)d_in[4];
    const float* kw   = (const float*)d_in[5];
    const float* vw   = (const float*)d_in[6];
    const float* ow   = (const float*)d_in[7];
    const float* qnw  = (const float*)d_in[8];
    const float* knw  = (const float*)d_in[9];
    float* out = (float*)d_out;

    char* ws = (char*)d_ws;
    __bf16* hid_bf = (__bf16*)ws; ws += (size_t)MROWS * HDIM * 2;
    __bf16* wqkv   = (__bf16*)ws; ws += (size_t)QKVN * HDIM * 2;
    __bf16* wo     = (__bf16*)ws; ws += (size_t)HDIM * HDIM * 2;
    float*  qkv    = (float*)ws;  ws += (size_t)MROWS * QKVN * 4;
    __bf16* Qb     = (__bf16*)ws; ws += (size_t)BSZ * NH  * SLEN * HD * 2;
    __bf16* Kb     = (__bf16*)ws; ws += (size_t)BSZ * KVH * SLEN * HD * 2;
    __bf16* Vt     = (__bf16*)ws; ws += (size_t)BSZ * KVH * HD * SLEN * 2;
    __bf16* AO     = (__bf16*)ws; ws += (size_t)MROWS * HDIM * 2;

    f32_to_bf16_k<<<(MROWS * HDIM + 255) / 256, 256, 0, stream>>>(hidden, hid_bf, MROWS * HDIM);
    f32_to_bf16_k<<<(1024 * 1024 + 255) / 256, 256, 0, stream>>>(qw, wqkv, 1024 * 1024);
    f32_to_bf16_k<<<(256 * 1024 + 255) / 256, 256, 0, stream>>>(kw, wqkv + 1024 * 1024, 256 * 1024);
    f32_to_bf16_k<<<(256 * 1024 + 255) / 256, 256, 0, stream>>>(vw, wqkv + 1280 * 1024, 256 * 1024);
    f32_to_bf16_k<<<(1024 * 1024 + 255) / 256, 256, 0, stream>>>(ow, wo, 1024 * 1024);

    // fused QKV projection: [4096,1536] = hid_bf @ wqkv^T  (block tile 32x256)
    dim3 g1(QKVN / 256, MROWS / 32);
    gemm_bf16_nt<<<g1, 128, 0, stream>>>(hid_bf, wqkv, qkv, MROWS, QKVN, HDIM);

    qkv_post_k<<<(BSZ * SLEN * 24) / 4, 128, 0, stream>>>(qkv, cosb, sinb, qnw, knw, Qb, Kb, Vt);

    dim3 g2(SLEN / 64, NH, BSZ);
    flash_attn_k<<<g2, 128, 0, stream>>>(Qb, Kb, Vt, AO);

    // output projection: [4096,1024] f32 = AO @ wo^T
    dim3 g3(HDIM / 256, MROWS / 32);
    gemm_bf16_nt<<<g3, 128, 0, stream>>>(AO, wo, out, MROWS, HDIM, HDIM);
}
